// MTLModel_63110249447856
// MI455X (gfx1250) — matrix operand order repaired and checked by reading, not compiled, not run on hardware
//
#include <hip/hip_runtime.h>
#include <hip/hip_bf16.h>
#include <stdint.h>

// ---------------------------------------------------------------------------
// CDNA5 / gfx1250: segment-mean -> BiGRU x2 -> FF -> head.
// GEMMs on v_wmma_f32_16x16x32_bf16 (fp32 accumulate), A-tiles staged with
// gfx1250 async global->LDS copies (ASYNCcnt), software-pipelined K loop.
// ---------------------------------------------------------------------------

typedef __bf16 bf16;
typedef float  v8f   __attribute__((ext_vector_type(8)));
typedef bf16   v16bf __attribute__((ext_vector_type(16)));

#define B_BATCH 512
#define T_TOK   128
#define D_EMB   768
#define S_SEG   64
#define BS_ROWS (B_BATCH * S_SEG)   // 32768 GEMM rows

// -------- gfx1250 async global->LDS copy (ISA 10.x / 15.18.3, ASYNCcnt) ----
__device__ __forceinline__ void async_g2l_b128(uint32_t lds_off, const void* g) {
  asm volatile("global_load_async_to_lds_b128 %0, %1, off"
               :: "v"(lds_off), "v"((uint64_t)(uintptr_t)g)
               : "memory");
}
__device__ __forceinline__ void wait_async0() {
  asm volatile("s_wait_asynccnt 0x0" ::: "memory");
}

// ---------------------------------------------------------------------------
__global__ void f32_to_bf16_k(const float* __restrict__ src,
                              bf16* __restrict__ dst, int n) {
  int i = blockIdx.x * blockDim.x + threadIdx.x;
  if (i < n) dst[i] = (bf16)src[i];
}

// ---------------------------------------------------------------------------
// Segment mean: word_ids sorted per row; one block per batch row.
// ---------------------------------------------------------------------------
__global__ __launch_bounds__(256)
void segment_mean_k(const float* __restrict__ emb,
                    const int* __restrict__ wid_g,
                    float* __restrict__ avg) {
  __shared__ int wid[T_TOK];
  const int b = blockIdx.x;
  const int tid = threadIdx.x;
  if (tid < T_TOK) wid[tid] = wid_g[b * T_TOK + tid];
  __syncthreads();

  float* arow = avg + (size_t)b * S_SEG * D_EMB;
  for (int i = tid; i < S_SEG * D_EMB; i += 256) arow[i] = 0.f;
  __syncthreads();   // order zero-fill before mean writes (same block)

  const float* erow = emb + (size_t)b * T_TOK * D_EMB;
  for (int c = 0; c < 3; ++c) {            // 768 cols / 256 threads
    const int d = tid + c * 256;
    float sum = 0.f, cnt = 0.f;
    int cur = wid[0];
    for (int t = 0; t < T_TOK; ++t) {
      const int w = wid[t];
      if (w != cur) {
        arow[cur * D_EMB + d] = sum / cnt;
        sum = 0.f; cnt = 0.f; cur = w;
      }
      sum += erow[t * D_EMB + d];
      cnt += 1.f;
    }
    arow[cur * D_EMB + d] = sum / cnt;
  }
}

// ---------------------------------------------------------------------------
// WMMA GEMM: C[M,N] = A[M,K](f32) * W[N,K](bf16)^T + bias, optional ReLU.
// Block = 256 thr (8 waves), tile 64(M) x 128(N), K-chunk 32, K/32 even.
// A-tiles: async global->LDS (fp32), double buffered; B fragments: two 16B
// global loads at k 8h..8h+7 and 16+8h.. (W stored N x K), ping-pong prefetched.
// ---------------------------------------------------------------------------
__global__ __launch_bounds__(256)
void gemm_bf16_wmma(const float* __restrict__ A,
                    const bf16* __restrict__ W,
                    const float* __restrict__ bias,
                    float* __restrict__ C,
                    int M, int N, int K, int relu) {
  __shared__ float Af[2][64][36];          // fp32 A tiles, +4 pad (144B rows)
  const int tid  = threadIdx.x;
  const int wave = tid >> 5;
  const int lane = tid & 31;
  const int l15  = lane & 15;
  const int lhi  = lane >> 4;
  const int mBase = blockIdx.x * 64;
  const int nBase = blockIdx.y * 128;
  const int mt  = wave & 3;                // 4 M-subtiles
  const int ntb = (wave >> 2) * 4;         // 2 wave-groups x 4 N-subtiles

  const uint32_t ldsAf = (uint32_t)(uintptr_t)&Af[0][0][0];
  const int arow = tid >> 2;               // staging row 0..63
  const int aseg = tid & 3;                // 16B segment; +4 = second segment

  v8f acc[4] = {};

  // stage one 64x32 fp32 A chunk into LDS buffer `buf` via async DMA
  auto stageA = [&](int k0, int buf) {
    const float* g = A + (size_t)(mBase + arow) * K + k0 + aseg * 4;
    uint32_t l = ldsAf + (uint32_t)(((buf * 64 + arow) * 36 + aseg * 4) * 4);
    async_g2l_b128(l, g);
    async_g2l_b128(l + 64, g + 16);        // segment aseg+4
  };
  // load 4 B fragments (two 16B loads each: k 8h.., 16+8h..) for k-chunk k0
  auto loadB = [&](int k0, v16bf* bf) {
    typedef bf16 v8bf_ __attribute__((ext_vector_type(8)));
#pragma unroll
    for (int j = 0; j < 4; ++j) {
      const int col = nBase + (ntb + j) * 16 + l15;
      const v8bf_ lo = *(const v8bf_*)(W + (size_t)col * K + k0 + lhi * 8);
      const v8bf_ hi = *(const v8bf_*)(W + (size_t)col * K + k0 + 16 + lhi * 8);
      for (int i = 0; i < 8; ++i) { bf[j][i] = lo[i]; bf[j][8 + i] = hi[i]; }
    }
  };
  // build bf16 A fragment from fp32 LDS buffer + 4 WMMAs
  auto compute = [&](int buf, const v16bf* bf) {
    const float* r = &Af[buf][mt * 16 + l15][0];
    const float4 f0 = *(const float4*)(r + lhi * 8);
    const float4 f1 = *(const float4*)(r + lhi * 8 + 4);
    const float4 f2 = *(const float4*)(r + 16 + lhi * 8);
    const float4 f3 = *(const float4*)(r + 16 + lhi * 8 + 4);
    v16bf a;
    a[0]=(bf16)f0.x; a[1]=(bf16)f0.y; a[2]=(bf16)f0.z; a[3]=(bf16)f0.w;
    a[4]=(bf16)f1.x; a[5]=(bf16)f1.y; a[6]=(bf16)f1.z; a[7]=(bf16)f1.w;
    a[8]=(bf16)f2.x; a[9]=(bf16)f2.y; a[10]=(bf16)f2.z; a[11]=(bf16)f2.w;
    a[12]=(bf16)f3.x; a[13]=(bf16)f3.y; a[14]=(bf16)f3.z; a[15]=(bf16)f3.w;
#pragma unroll
    for (int j = 0; j < 4; ++j)
      acc[j] = __builtin_amdgcn_wmma_f32_16x16x32_bf16(
          false, a, false, bf[j], (short)0, acc[j], false, false);
  };

  v16bf bA[4], bB[4];
  const int NK = K >> 5;                   // 24 / 16 / 16 : always even
  stageA(0, 0);
  loadB(0, bA);
  wait_async0();
  __syncthreads();

  for (int kc = 0; kc < NK; kc += 2) {
    // phase 0: compute chunk kc, prefetch kc+1
    stageA((kc + 1) << 5, 1);
    loadB((kc + 1) << 5, bB);
    compute(0, bA);
    wait_async0();
    __syncthreads();
    // phase 1: compute chunk kc+1, prefetch kc+2
    if (kc + 2 < NK) {
      stageA((kc + 2) << 5, 0);
      loadB((kc + 2) << 5, bA);
    }
    compute(1, bB);
    wait_async0();
    __syncthreads();
  }

#pragma unroll
  for (int j = 0; j < 4; ++j) {
    const int n = nBase + (ntb + j) * 16 + l15;
    const float bv = bias[n];
#pragma unroll
    for (int v = 0; v < 8; ++v) {
      const int m = mBase + mt * 16 + lhi * 8 + v;
      float val = acc[j][v] + bv;
      if (relu) val = fmaxf(val, 0.f);
      C[(size_t)m * N + n] = val;
    }
  }
}

// ---------------------------------------------------------------------------
// GRU recurrence (one layer; blockIdx.y = direction).
// Block owns 16 batch rows for all 64 steps; h in LDS (fp32 + bf16 mirror).
// Per step: gh[16,768] = h @ w_hh^T via WMMA (B frags ping-pong prefetched
// from L2-resident w_hh), then fp32 gate math.
// ---------------------------------------------------------------------------
__global__ __launch_bounds__(256)
void gru_layer_k(const float* __restrict__ gi,   // [B][64][1536] (dir*768)
                 const bf16* __restrict__ whh,   // [2][768][256]
                 const float* __restrict__ bhh,  // [2][768]
                 float* __restrict__ y) {        // [B][64][512]
  __shared__ float hF[16][256];
  __shared__ bf16  hB[16][264];                  // +8 pad
  __shared__ float gh[16][768];

  const int tid  = threadIdx.x;
  const int wave = tid >> 5;
  const int lane = tid & 31;
  const int l15  = lane & 15;
  const int lhi  = lane >> 4;
  const int b0   = blockIdx.x * 16;
  const int dir  = blockIdx.y;
  const bf16*  Wd = whh + (size_t)dir * 768 * 256;
  const float* bh = bhh + dir * 768;

  for (int i = tid; i < 16 * 256; i += 256) {
    hF[i >> 8][i & 255] = 0.f;
    hB[i >> 8][i & 255] = (bf16)0.f;
  }
  __syncthreads();

  typedef bf16 v8bf __attribute__((ext_vector_type(8)));

  for (int step = 0; step < 64; ++step) {
    const int t = dir ? (63 - step) : step;

    // gh = h @ w_hh^T : M=16, N=768 (wave owns 6 N-subtiles), K=256
    v8f acc[6] = {};
    v16bf bfr[2][6];
#pragma unroll
    for (int j = 0; j < 6; ++j) {
      const int col = (wave * 6 + j) * 16 + l15;
      const v8bf lo = *(const v8bf*)(Wd + (size_t)col * 256 + lhi * 8);
      const v8bf hi = *(const v8bf*)(Wd + (size_t)col * 256 + 16 + lhi * 8);
      for (int i = 0; i < 8; ++i) { bfr[0][j][i] = lo[i]; bfr[0][j][8 + i] = hi[i]; }
    }
#pragma unroll
    for (int kc = 0; kc < 8; ++kc) {
      const int k0 = kc * 32;
      const v8bf alo = *(const v8bf*)&hB[l15][k0 + lhi * 8];
      const v8bf ahi = *(const v8bf*)&hB[l15][k0 + 16 + lhi * 8];
      v16bf afrag;
#pragma unroll
      for (int i = 0; i < 8; ++i) { afrag[i] = alo[i]; afrag[8 + i] = ahi[i]; }
      if (kc < 7) {
#pragma unroll
        for (int j = 0; j < 6; ++j) {
          const int col = (wave * 6 + j) * 16 + l15;
          const v8bf lo = *(const v8bf*)(Wd + (size_t)col * 256 + (k0 + 32) + lhi * 8);
          const v8bf hi = *(const v8bf*)(Wd + (size_t)col * 256 + (k0 + 48) + lhi * 8);
          for (int i = 0; i < 8; ++i) { bfr[(kc + 1) & 1][j][i] = lo[i]; bfr[(kc + 1) & 1][j][8 + i] = hi[i]; }
        }
      }
#pragma unroll
      for (int j = 0; j < 6; ++j)
        acc[j] = __builtin_amdgcn_wmma_f32_16x16x32_bf16(
            false, afrag, false, bfr[kc & 1][j], (short)0, acc[j], false, false);
    }
#pragma unroll
    for (int j = 0; j < 6; ++j) {
      const int n = (wave * 6 + j) * 16 + l15;
#pragma unroll
      for (int v = 0; v < 8; ++v) gh[lhi * 8 + v][n] = acc[j][v];
    }
    __syncthreads();

    // fp32 gate math (r, z, n gate order)
    for (int e = tid; e < 16 * 256; e += 256) {
      const int b = e >> 8, j = e & 255;
      const size_t gbase =
          ((size_t)(b0 + b) * S_SEG + t) * 1536 + (size_t)dir * 768 + j;
      const float gr = gi[gbase];
      const float gz = gi[gbase + 256];
      const float gn = gi[gbase + 512];
      const float hr = gh[b][j]       + bh[j];
      const float hz = gh[b][256 + j] + bh[256 + j];
      const float hn = gh[b][512 + j] + bh[512 + j];
      const float r = 1.f / (1.f + __expf(-(gr + hr)));
      const float z = 1.f / (1.f + __expf(-(gz + hz)));
      const float n = tanhf(gn + r * hn);
      const float h = (1.f - z) * n + z * hF[b][j];
      hF[b][j] = h;
      hB[b][j] = (bf16)h;
      y[((size_t)(b0 + b) * S_SEG + t) * 512 + dir * 256 + j] = h;
    }
    __syncthreads();
  }
}

// ---------------------------------------------------------------------------
__global__ void out_proj_k(const float* __restrict__ feats,
                           const float* __restrict__ ow,
                           const float* __restrict__ ob,
                           float* __restrict__ out) {
  const int i = blockIdx.x * blockDim.x + threadIdx.x;
  if (i >= B_BATCH * 2 * S_SEG) return;
  const int s = i & 63;
  const int c = (i >> 6) & 1;
  const int b = i >> 7;
  const float* f = feats + ((size_t)b * S_SEG + s) * 512;
  const float* w = ow + c * 512;
  float acc = ob[c];
  for (int k = 0; k < 512; ++k) acc += f[k] * w[k];
  out[((size_t)b * 2 + c) * S_SEG + s] = acc;
}

// ---------------------------------------------------------------------------
static inline size_t align256(size_t x) { return (x + 255) & ~(size_t)255; }

extern "C" void kernel_launch(void* const* d_in, const int* in_sizes, int n_in,
                              void* d_out, int out_size, void* d_ws, size_t ws_size,
                              hipStream_t stream) {
  const float* emb    = (const float*)d_in[0];
  const int*   wid    = (const int*)  d_in[1];
  const float* w_ih0  = (const float*)d_in[2];   // [2][768][768] = [1536][768]
  const float* w_hh0  = (const float*)d_in[3];   // [2][768][256]
  const float* b_ih0  = (const float*)d_in[4];   // [1536]
  const float* b_hh0  = (const float*)d_in[5];
  const float* w_ih1  = (const float*)d_in[6];   // [1536][512]
  const float* w_hh1  = (const float*)d_in[7];
  const float* b_ih1  = (const float*)d_in[8];
  const float* b_hh1  = (const float*)d_in[9];
  const float* ff_w   = (const float*)d_in[10];  // [512][512]
  const float* ff_b   = (const float*)d_in[11];
  const float* out_w  = (const float*)d_in[12];  // [2][512]
  const float* out_b  = (const float*)d_in[13];
  float* out = (float*)d_out;

  char* ws = (char*)d_ws;
  size_t o = 0;
  bf16* wih0_bf = (bf16*)(ws + o); o = align256(o + (size_t)1536 * 768 * 2);
  bf16* whh0_bf = (bf16*)(ws + o); o = align256(o + (size_t)2 * 768 * 256 * 2);
  bf16* wih1_bf = (bf16*)(ws + o); o = align256(o + (size_t)1536 * 512 * 2);
  bf16* whh1_bf = (bf16*)(ws + o); o = align256(o + (size_t)2 * 768 * 256 * 2);
  bf16* ffw_bf  = (bf16*)(ws + o); o = align256(o + (size_t)512 * 512 * 2);
  float* bufA = (float*)(ws + o); o += (size_t)BS_ROWS * 768 * 4;   // avg, then h1
  float* bufB = (float*)(ws + o); o += (size_t)BS_ROWS * 1536 * 4;  // gi0/gi1
  float* bufC = (float*)(ws + o); o += (size_t)BS_ROWS * 512 * 4;   // h0, then feats

  auto conv = [&](const float* s, bf16* d, int n) {
    f32_to_bf16_k<<<(n + 255) / 256, 256, 0, stream>>>(s, d, n);
  };
  conv(w_ih0, wih0_bf, 1536 * 768);
  conv(w_hh0, whh0_bf, 2 * 768 * 256);
  conv(w_ih1, wih1_bf, 1536 * 512);
  conv(w_hh1, whh1_bf, 2 * 768 * 256);
  conv(ff_w,  ffw_bf,  512 * 512);

  segment_mean_k<<<B_BATCH, 256, 0, stream>>>(emb, wid, bufA);

  gemm_bf16_wmma<<<dim3(BS_ROWS / 64, 1536 / 128), 256, 0, stream>>>(
      bufA, wih0_bf, b_ih0, bufB, BS_ROWS, 1536, 768, 0);

  gru_layer_k<<<dim3(B_BATCH / 16, 2), 256, 0, stream>>>(bufB, whh0_bf, b_hh0, bufC);

  gemm_bf16_wmma<<<dim3(BS_ROWS / 64, 1536 / 128), 256, 0, stream>>>(
      bufC, wih1_bf, b_ih1, bufB, BS_ROWS, 1536, 512, 0);

  gru_layer_k<<<dim3(B_BATCH / 16, 2), 256, 0, stream>>>(bufB, whh1_bf, b_hh1, bufA);

  gemm_bf16_wmma<<<dim3(BS_ROWS / 64, 512 / 128), 256, 0, stream>>>(
      bufA, ffw_bf, ff_b, bufC, BS_ROWS, 512, 512, 1);

  out_proj_k<<<(B_BATCH * 2 * S_SEG + 255) / 256, 256, 0, stream>>>(
      bufC, out_w, out_b, out);
}
